// HiViT_28028956573914
// MI455X (gfx1250) — matrix operand order repaired and checked by reading, not compiled, not run on hardware
//
#include <hip/hip_runtime.h>
#include <hip/hip_bf16.h>

// ---------------------------------------------------------------------------
// HiViT forward for MI455X (gfx1250, wave32, WMMA).
// bf16 WMMA inputs, f32 accumulate (v_wmma_f32_16x16x32_bf16).
// GEMM: 32(M)x128(N) block tile, 8 waves x (16x32) wave tiles, LDS tiles
// stored in fragment order -> 2x ds_load_b128 per fragment.
// Attention: N=196 padded to 224; float4 global fragment loads.
// Workspace peak: 16,056,320 floats (~64.2 MB).
// ---------------------------------------------------------------------------

#define B_    8
#define NP    196
#define EMB_  512
#define NH    8
#define HD    64
#define NPAD  224
#define D4_   128
#define D8_   256
#define RPEL  729

typedef __attribute__((ext_vector_type(16))) __bf16 v16bf;
typedef __attribute__((ext_vector_type(8)))  float  v8f;

__device__ __forceinline__ __bf16 to_bf(float f) { return (__bf16)f; }
__device__ __forceinline__ unsigned pack2(float a, float b) {
    union { __bf16 h[2]; unsigned u; } x;
    x.h[0] = (__bf16)a; x.h[1] = (__bf16)b; return x.u;
}
__device__ __forceinline__ float gelu_f(float x) {
    return 0.5f * x * (1.0f + erff(x * 0.70710678118654752f));
}
__device__ __forceinline__ float warp_sum(float x) {
    for (int o = 16; o > 0; o >>= 1) x += __shfl_xor(x, o, 32);
    return x;
}
__device__ __forceinline__ float warp_max(float x) {
    for (int o = 16; o > 0; o >>= 1) x = fmaxf(x, __shfl_xor(x, o, 32));
    return x;
}

// ---------------------------------------------------------------------------
// Patch embed: 4x4/4 conv (3->128) rearranged to [B, 196, 4,4, 128].
// ---------------------------------------------------------------------------
__global__ __launch_bounds__(256) void conv_patch_kernel(
    const float* __restrict__ img, const float* __restrict__ cw,
    const float* __restrict__ cb, float* __restrict__ out) {
    int idx = blockIdx.x * 256 + threadIdx.x;           // B*196*16*128
    if (idx >= B_ * NP * 16 * D4_) return;
    int c = idx & 127;
    int t = idx >> 7;
    int inner = t & 15; t >>= 4;
    int n = t % NP;
    int b = t / NP;
    int ii = inner >> 2, jj = inner & 3;
    int hp = n / 14, wp = n % 14;
    int h = hp * 4 + ii, w = wp * 4 + jj;               // conv-out coords (56x56)
    float acc = cb[c];
    for (int ic = 0; ic < 3; ic++)
        for (int kh = 0; kh < 4; kh++)
            for (int kw = 0; kw < 4; kw++)
                acc += img[((b * 3 + ic) * 224 + h * 4 + kh) * 224 + w * 4 + kw] *
                       cw[((c * 3 + ic) * 4 + kh) * 4 + kw];
    out[idx] = acc;
}

// ---------------------------------------------------------------------------
// LayerNorm over last dim D (multiple of 32, <=1024). Wave per row.
// ---------------------------------------------------------------------------
__global__ __launch_bounds__(256) void layernorm_kernel(
    const float* __restrict__ x, float* __restrict__ y,
    const float* __restrict__ g, const float* __restrict__ b,
    int rows, int D) {
    int wid = threadIdx.x >> 5, lane = threadIdx.x & 31;
    int row = blockIdx.x * 8 + wid;
    if (row >= rows) return;
    int cnt = D >> 5;
    float v[32];
    float s = 0.f;
    for (int i = 0; i < cnt; i++) {
        v[i] = x[(long)row * D + i * 32 + lane];
        s += v[i];
    }
    float mean = warp_sum(s) * (1.0f / D);
    float vs = 0.f;
    for (int i = 0; i < cnt; i++) { float d = v[i] - mean; vs += d * d; }
    float rstd = rsqrtf(warp_sum(vs) * (1.0f / D) + 1e-5f);
    for (int i = 0; i < cnt; i++) {
        int c = i * 32 + lane;
        y[(long)row * D + c] = (v[i] - mean) * rstd * g[c] + b[c];
    }
}

// ---------------------------------------------------------------------------
// GEMM: out[M,N] = act(A[M,K] @ W[K,N] + bias) (+ res).
// Block 256 thr = 8 waves; block tile 32x128; wave tile 16x32 (2 acc).
// LDS tiles stored in WMMA fragment order: lane reads v16bf = 2x b128.
//   A frag elem e (lane hi/lo):  K = e + 8*hi + 8*(e>=8)  (8-contig groups)
//   B frag elem e:               K = e + 8*hi + 8*(e>=8)  (same as A)
// act: 0=none, 1=gelu.
// ---------------------------------------------------------------------------
__global__ __launch_bounds__(256) void gemm_wmma_kernel(
    const float* __restrict__ A, const float* __restrict__ W,
    const float* __restrict__ bias, const float* __restrict__ res,
    float* __restrict__ out, int M, int N, int K, int act) {
    __shared__ __align__(32) __bf16 sAf[2][32][16];   // [mblk][lane][e]
    __shared__ __align__(32) __bf16 sBf[8][32][16];   // [n16blk][lane][e]
    int tid = threadIdx.x;
    int wv = tid >> 5, lane = tid & 31;
    int hi = lane >> 4, lo = lane & 15;
    int m0 = blockIdx.x * 32;
    int n0 = blockIdx.y * 128;
    int mi = wv & 1;             // wave's 16-row half
    int nj = wv >> 1;            // wave's 32-col group (0..3)

    // staging coords (fixed per thread)
    int sc = tid & 127;          // B col within 128
    int srb = (tid >> 7) * 2;    // B starting row pair (0 or 2)
    unsigned* bDst = (unsigned*)&sBf[sc >> 4][(sc & 15)][0]; // + hi*16 row sel below

    v8f acc0, acc1;
    for (int i = 0; i < 8; i++) { acc0[i] = 0.0f; acc1[i] = 0.0f; }

    for (int k0 = 0; k0 < K; k0 += 32) {
        // ---- stage A tile (32 rows x 32 K), fragment order, packed pairs ----
        for (int s = 0; s < 2; s++) {
            int p = tid + 256 * s;            // pair index 0..511
            int row = p >> 4;                 // 0..31
            int kp = (p & 15) * 2;            // even K
            float2 av = *(const float2*)&A[(long)(m0 + row) * K + k0 + kp];
            int h2 = (kp >> 3) & 1;
            int e0 = kp - 8 * h2 - ((kp >= 16) ? 8 : 0);
            ((unsigned*)&sAf[row >> 4][h2 * 16 + (row & 15)][0])[e0 >> 1] =
                pack2(av.x, av.y);
        }
        // ---- stage W tile (32 K x 128 N), fragment order, packed pairs ----
        for (int s = 0; s < 8; s++) {
            int r = srb + 4 * s;              // even row; (r, r+1) same half
            float w0 = W[(long)(k0 + r) * N + n0 + sc];
            float w1 = W[(long)(k0 + r + 1) * N + n0 + sc];
            int h2 = (r >> 3) & 1;
            ((unsigned*)&sBf[sc >> 4][h2 * 16 + (sc & 15)][0])[(r - 8 * h2 - ((r >= 16) ? 8 : 0)) >> 1] =
                pack2(w0, w1);
        }
        if (k0 + 32 < K) {
            __builtin_prefetch(&W[(long)(k0 + 32 + srb) * N + n0 + sc], 0, 1);
            __builtin_prefetch(&A[(long)(m0 + (tid >> 4)) * K + k0 + 32], 0, 1);
        }
        __syncthreads();

        v16bf af  = *(const v16bf*)&sAf[mi][lane][0];
        v16bf bf0 = *(const v16bf*)&sBf[nj * 2][lane][0];
        v16bf bf1 = *(const v16bf*)&sBf[nj * 2 + 1][lane][0];
        acc0 = __builtin_amdgcn_wmma_f32_16x16x32_bf16(
            false, af, false, bf0, (short)0, acc0, false, false);
        acc1 = __builtin_amdgcn_wmma_f32_16x16x32_bf16(
            false, af, false, bf1, (short)0, acc1, false, false);
        __syncthreads();
    }

    int colA = n0 + nj * 32 + lo;
    int colB = colA + 16;
    float bvA = bias ? bias[colA] : 0.0f;
    float bvB = bias ? bias[colB] : 0.0f;
    float va[8], vb[8];
    for (int r = 0; r < 8; r++) {
        va[r] = acc0[r] + bvA;
        vb[r] = acc1[r] + bvB;
        if (act == 1) { va[r] = gelu_f(va[r]); vb[r] = gelu_f(vb[r]); }
    }
    int rowb = m0 + mi * 16 + (hi ? 8 : 0);
    if (res) {
        for (int r = 0; r < 8; r++) {
            va[r] += res[(long)(rowb + r) * N + colA];
            vb[r] += res[(long)(rowb + r) * N + colB];
        }
    }
    for (int r = 0; r < 8; r++) {
        out[(long)(rowb + r) * N + colA] = va[r];
        out[(long)(rowb + r) * N + colB] = vb[r];
    }
}

// ---------------------------------------------------------------------------
// Attention scores: S[bh,224,224] = (Q @ K^T)*0.125 + rpe_bias, -1e30 pad.
// Wave per 16x16 tile; K-dim 64 (2 WMMA). float4 fragment loads.
// ---------------------------------------------------------------------------
__global__ __launch_bounds__(32) void attn_scores_kernel(
    const float* __restrict__ qkv, const float* __restrict__ tab,
    const int* __restrict__ rpe, float* __restrict__ S) {
    int mt = blockIdx.x, nt = blockIdx.y, bh = blockIdx.z;
    int b = bh / NH, h = bh % NH;
    int lane = threadIdx.x;
    int hi = lane >> 4, lo = lane & 15;

    v8f acc;
    for (int i = 0; i < 8; i++) acc[i] = 0.0f;

    int qrow = mt * 16 + lo;
    int kcol = nt * 16 + lo;
    const float* qp = qkv + (long)(b * NP + (qrow < NP ? qrow : 0)) * 1536 + h * HD;
    const float* kp = qkv + (long)(b * NP + (kcol < NP ? kcol : 0)) * 1536 + EMB_ + h * HD;
    bool qv = qrow < NP, kv = kcol < NP;

    for (int kk = 0; kk < HD; kk += 32) {
        v16bf af, bfr;
        if (qv) {
            const float* g = qp + kk + hi * 8;
            float4 a0 = *(const float4*)(g);
            float4 a1 = *(const float4*)(g + 4);
            float4 a2 = *(const float4*)(g + 16);
            float4 a3 = *(const float4*)(g + 20);
            af[0]=to_bf(a0.x); af[1]=to_bf(a0.y); af[2]=to_bf(a0.z); af[3]=to_bf(a0.w);
            af[4]=to_bf(a1.x); af[5]=to_bf(a1.y); af[6]=to_bf(a1.z); af[7]=to_bf(a1.w);
            af[8]=to_bf(a2.x); af[9]=to_bf(a2.y); af[10]=to_bf(a2.z); af[11]=to_bf(a2.w);
            af[12]=to_bf(a3.x); af[13]=to_bf(a3.y); af[14]=to_bf(a3.z); af[15]=to_bf(a3.w);
        } else for (int e = 0; e < 16; e++) af[e] = to_bf(0.0f);
        if (kv) {
            const float* g = kp + kk + hi * 8;
            float4 b0 = *(const float4*)(g);
            float4 b1 = *(const float4*)(g + 4);
            float4 b2 = *(const float4*)(g + 16);
            float4 b3 = *(const float4*)(g + 20);
            bfr[0]=to_bf(b0.x); bfr[1]=to_bf(b0.y); bfr[2]=to_bf(b0.z); bfr[3]=to_bf(b0.w);
            bfr[4]=to_bf(b1.x); bfr[5]=to_bf(b1.y); bfr[6]=to_bf(b1.z); bfr[7]=to_bf(b1.w);
            bfr[8]=to_bf(b2.x); bfr[9]=to_bf(b2.y); bfr[10]=to_bf(b2.z); bfr[11]=to_bf(b2.w);
            bfr[12]=to_bf(b3.x); bfr[13]=to_bf(b3.y); bfr[14]=to_bf(b3.z); bfr[15]=to_bf(b3.w);
        } else for (int e = 0; e < 16; e++) bfr[e] = to_bf(0.0f);
        acc = __builtin_amdgcn_wmma_f32_16x16x32_bf16(
            false, af, false, bfr, (short)0, acc, false, false);
    }
    for (int r = 0; r < 8; r++) {
        int row = mt * 16 + r + (hi ? 8 : 0);
        int col = nt * 16 + lo;
        float s;
        if (row < NP && col < NP)
            s = acc[r] * 0.125f + tab[rpe[row * NP + col] * NH + h];
        else
            s = -1e30f;
        S[((long)bh * NPAD + row) * NPAD + col] = s;
    }
}

// ---------------------------------------------------------------------------
// Softmax over 224 columns; wave per row.
// ---------------------------------------------------------------------------
__global__ __launch_bounds__(256) void softmax224_kernel(
    const float* __restrict__ S, float* __restrict__ P) {
    int wid = threadIdx.x >> 5, lane = threadIdx.x & 31;
    long row = (long)blockIdx.x * 8 + wid;    // B*NH*NPAD rows
    float v[7];
    float mx = -3.4e38f;
    for (int i = 0; i < 7; i++) {
        v[i] = S[row * NPAD + i * 32 + lane];
        mx = fmaxf(mx, v[i]);
    }
    mx = warp_max(mx);
    float sum = 0.f;
    for (int i = 0; i < 7; i++) { v[i] = expf(v[i] - mx); sum += v[i]; }
    float inv = 1.0f / warp_sum(sum);
    for (int i = 0; i < 7; i++)
        P[row * NPAD + i * 32 + lane] = v[i] * inv;
}

// ---------------------------------------------------------------------------
// attn out: y[b,row,h*64+col] = P[bh] @ V. K-dim 224 (7 WMMA steps).
// ---------------------------------------------------------------------------
__global__ __launch_bounds__(32) void attn_out_kernel(
    const float* __restrict__ P, const float* __restrict__ qkv,
    float* __restrict__ y) {
    int mt = blockIdx.x, nt = blockIdx.y, bh = blockIdx.z;
    int b = bh / NH, h = bh % NH;
    int lane = threadIdx.x;
    int hi = lane >> 4, lo = lane & 15;

    v8f acc;
    for (int i = 0; i < 8; i++) acc[i] = 0.0f;

    int prow = mt * 16 + lo;          // < 208 < NPAD, always valid
    int vcol = nt * 16 + lo;          // < 64
    const float* pp = P + ((long)bh * NPAD + prow) * NPAD;
    for (int kk = 0; kk < NPAD; kk += 32) {
        v16bf af, bfr;
        const float* g = pp + kk + hi * 8;
        float4 a0 = *(const float4*)(g);
        float4 a1 = *(const float4*)(g + 4);
        float4 a2 = *(const float4*)(g + 16);
        float4 a3 = *(const float4*)(g + 20);
        af[0]=to_bf(a0.x); af[1]=to_bf(a0.y); af[2]=to_bf(a0.z); af[3]=to_bf(a0.w);
        af[4]=to_bf(a1.x); af[5]=to_bf(a1.y); af[6]=to_bf(a1.z); af[7]=to_bf(a1.w);
        af[8]=to_bf(a2.x); af[9]=to_bf(a2.y); af[10]=to_bf(a2.z); af[11]=to_bf(a2.w);
        af[12]=to_bf(a3.x); af[13]=to_bf(a3.y); af[14]=to_bf(a3.z); af[15]=to_bf(a3.w);
        for (int e = 0; e < 16; e++) {
            int kb = kk + hi * 8 + e + ((e >= 8) ? 8 : 0);
            bfr[e] = (kb < NP)
                ? to_bf(qkv[((long)(b * NP + kb)) * 1536 + 2 * EMB_ + h * HD + vcol])
                : to_bf(0.0f);
        }
        acc = __builtin_amdgcn_wmma_f32_16x16x32_bf16(
            false, af, false, bfr, (short)0, acc, false, false);
    }
    for (int r = 0; r < 8; r++) {
        int row = mt * 16 + r + (hi ? 8 : 0);
        if (row < NP)
            y[((long)(b * NP + row)) * EMB_ + h * HD + nt * 16 + lo] = acc[r];
    }
}

// ---------------------------------------------------------------------------
// Patch-merge gathers, APE add, output copy.
// ---------------------------------------------------------------------------
__global__ __launch_bounds__(256) void merge1_gather_kernel(
    const float* __restrict__ x, float* __restrict__ o) {
    int idx = blockIdx.x * 256 + threadIdx.x;       // 6272*512
    if (idx >= 6272 * 512) return;
    int c512 = idx & 511;
    int r = idx >> 9;                               // [b][n][i2][j2]
    int g = c512 >> 7, c = c512 & 127;
    int j2 = r & 1, i2 = (r >> 1) & 1;
    int bn = r >> 2;                                // b*196+n
    int ii = 2 * i2 + (g & 1);
    int jj = 2 * j2 + (g >> 1);
    o[idx] = x[(((long)bn * 4 + ii) * 4 + jj) * D4_ + c];
}

__global__ __launch_bounds__(256) void merge2_gather_kernel(
    const float* __restrict__ x, float* __restrict__ o) {
    int idx = blockIdx.x * 256 + threadIdx.x;       // 1568*1024
    if (idx >= 1568 * 1024) return;
    int c1024 = idx & 1023;
    int r = idx >> 10;                              // b*196+n
    int g = c1024 >> 8, c = c1024 & 255;
    o[idx] = x[(((long)r * 2 + (g & 1)) * 2 + (g >> 1)) * D8_ + c];
}

__global__ __launch_bounds__(256) void add_ape_kernel(
    float* __restrict__ x, const float* __restrict__ ape) {
    int idx = blockIdx.x * 256 + threadIdx.x;       // 1568*512
    if (idx >= 1568 * EMB_) return;
    x[idx] += ape[idx % (NP * EMB_)];
}

__global__ __launch_bounds__(256) void copy_kernel(
    const float* __restrict__ x, float* __restrict__ o, int n) {
    int idx = blockIdx.x * 256 + threadIdx.x;
    if (idx < n) o[idx] = x[idx];
}

// ---------------------------------------------------------------------------
// Host orchestration.
// ---------------------------------------------------------------------------
static inline void gemm(const float* A, const float* W, const float* bias,
                        const float* res, float* out, int M, int N, int K,
                        int act, hipStream_t s) {
    dim3 g(M / 32, N / 128);
    gemm_wmma_kernel<<<g, 256, 0, s>>>(A, W, bias, res, out, M, N, K, act);
}
static inline void ln(const float* x, float* y, const float* g, const float* b,
                      int rows, int D, hipStream_t s) {
    layernorm_kernel<<<(rows + 7) / 8, 256, 0, s>>>(x, y, g, b, rows, D);
}

extern "C" void kernel_launch(void* const* d_in, const int* in_sizes, int n_in,
                              void* d_out, int out_size, void* d_ws, size_t ws_size,
                              hipStream_t stream) {
    const float* image   = (const float*)d_in[0];
    const float* conv_w  = (const float*)d_in[1];
    const float* conv_b  = (const float*)d_in[2];
    const float* pn_g    = (const float*)d_in[3];
    const float* pn_b    = (const float*)d_in[4];
    const float* s1_ng   = (const float*)d_in[5];
    const float* s1_nb   = (const float*)d_in[6];
    const float* s1_w1   = (const float*)d_in[7];
    const float* s1_b1   = (const float*)d_in[8];
    const float* s1_w2   = (const float*)d_in[9];
    const float* s1_b2   = (const float*)d_in[10];
    const float* s2_ng   = (const float*)d_in[11];
    const float* s2_nb   = (const float*)d_in[12];
    const float* s2_w1   = (const float*)d_in[13];
    const float* s2_b1   = (const float*)d_in[14];
    const float* s2_w2   = (const float*)d_in[15];
    const float* s2_b2   = (const float*)d_in[16];
    const float* pm1_ng  = (const float*)d_in[17];
    const float* pm1_nb  = (const float*)d_in[18];
    const float* pm1_rw  = (const float*)d_in[19];
    const float* pm2_ng  = (const float*)d_in[20];
    const float* pm2_nb  = (const float*)d_in[21];
    const float* pm2_rw  = (const float*)d_in[22];
    const float* ape     = (const float*)d_in[23];
    const float* mb_n1g  = (const float*)d_in[24];
    const float* mb_n1b  = (const float*)d_in[25];
    const float* mb_qkvw = (const float*)d_in[26];
    const float* mb_qkvb = (const float*)d_in[27];
    const float* mb_tab  = (const float*)d_in[28];
    const float* mb_projw= (const float*)d_in[29];
    const float* mb_projb= (const float*)d_in[30];
    const float* mb_n2g  = (const float*)d_in[31];
    const float* mb_n2b  = (const float*)d_in[32];
    const float* mb_w1   = (const float*)d_in[33];
    const float* mb_b1   = (const float*)d_in[34];
    const float* mb_w2   = (const float*)d_in[35];
    const float* mb_b2   = (const float*)d_in[36];
    const int*   rpe     = (const int*)d_in[37];

    float* ws = (float*)d_ws;
    // Region layout (floats): peak 16,056,320 (~64.2 MB)
    float* x0   = ws;                              // 3,211,264 (current activation)
    float* tmp1 = ws + 3211264;                    // 3,211,264 (LN scratch)
    float* big  = ws + 6422528;                    // 9,633,792 (hidden / qkv+S+P)
    float* qkv  = big;                             // 2,408,448
    float* S    = big + 2408448;                   // 3,211,264
    float* P    = S + 3211264;                     // 3,211,264
    float* yb   = tmp1 + 802816;                   // 802,816 (attn out)
    float* hidb = big;                             // 3,211,264 (block MLP hidden)

    // ---- Patch embed + patch-norm ----
    conv_patch_kernel<<<(B_ * NP * 16 * D4_) / 256, 256, 0, stream>>>(
        image, conv_w, conv_b, x0);
    ln(x0, x0, pn_g, pn_b, 25088, D4_, stream);

    // ---- Stage 1: 3 MLP layers over dim 128 ----
    for (int l = 0; l < 3; l++) {
        ln(x0, tmp1, s1_ng + l * D4_, s1_nb + l * D4_, 25088, D4_, stream);
        gemm(tmp1, s1_w1 + (long)l * D4_ * 384, s1_b1 + l * 384, nullptr,
             big, 25088, 384, D4_, 1, stream);
        gemm(big, s1_w2 + (long)l * 384 * D4_, s1_b2 + l * D4_, x0,
             x0, 25088, D4_, 384, 0, stream);
    }

    // ---- Merge 1: [B,196,4,4,128] -> [B,196,2,2,512] -> LN -> @rw -> 256 ----
    merge1_gather_kernel<<<(6272 * 512) / 256, 256, 0, stream>>>(x0, big);
    ln(big, big, pm1_ng, pm1_nb, 6272, 512, stream);
    gemm(big, pm1_rw, nullptr, nullptr, x0, 6272, D8_, 512, 0, stream);

    // ---- Stage 2: 3 MLP layers over dim 256 ----
    for (int l = 0; l < 3; l++) {
        ln(x0, tmp1, s2_ng + l * D8_, s2_nb + l * D8_, 6272, D8_, stream);
        gemm(tmp1, s2_w1 + (long)l * D8_ * 768, s2_b1 + l * 768, nullptr,
             big, 6272, 768, D8_, 1, stream);
        gemm(big, s2_w2 + (long)l * 768 * D8_, s2_b2 + l * D8_, x0,
             x0, 6272, D8_, 768, 0, stream);
    }

    // ---- Merge 2: -> [B,196,1024] -> LN -> @rw -> [B,196,512]; + APE ----
    merge2_gather_kernel<<<(1568 * 1024) / 256, 256, 0, stream>>>(x0, big);
    ln(big, big, pm2_ng, pm2_nb, 1568, 1024, stream);
    gemm(big, pm2_rw, nullptr, nullptr, x0, 1568, EMB_, 1024, 0, stream);
    add_ape_kernel<<<(1568 * EMB_) / 256, 256, 0, stream>>>(x0, ape);

    // ---- 24 transformer blocks ----
    for (int l = 0; l < 24; l++) {
        const float* qw = mb_qkvw + (long)l * EMB_ * 1536;
        const float* qb = mb_qkvb + (long)l * 1536;
        const float* tab= mb_tab  + (long)l * RPEL * NH;
        const float* pw = mb_projw+ (long)l * EMB_ * EMB_;
        const float* pb = mb_projb+ (long)l * EMB_;
        const float* w1 = mb_w1   + (long)l * EMB_ * 2048;
        const float* b1 = mb_b1   + (long)l * 2048;
        const float* w2 = mb_w2   + (long)l * 2048 * EMB_;
        const float* b2 = mb_b2   + (long)l * EMB_;

        ln(x0, tmp1, mb_n1g + l * EMB_, mb_n1b + l * EMB_, 1568, EMB_, stream);
        gemm(tmp1, qw, qb, nullptr, qkv, 1568, 1536, EMB_, 0, stream);

        attn_scores_kernel<<<dim3(14, 14, B_ * NH), 32, 0, stream>>>(
            qkv, tab, rpe, S);
        softmax224_kernel<<<(B_ * NH * NPAD) / 8, 256, 0, stream>>>(S, P);
        attn_out_kernel<<<dim3(13, 4, B_ * NH), 32, 0, stream>>>(P, qkv, yb);

        gemm(yb, pw, pb, x0, x0, 1568, EMB_, EMB_, 0, stream);

        ln(x0, tmp1, mb_n2g + l * EMB_, mb_n2b + l * EMB_, 1568, EMB_, stream);
        gemm(tmp1, w1, b1, nullptr, hidb, 1568, 2048, EMB_, 1, stream);
        gemm(hidb, w2, b2, x0, x0, 1568, EMB_, 2048, 0, stream);
    }

    copy_kernel<<<(1568 * EMB_ + 255) / 256, 256, 0, stream>>>(
        x0, (float*)d_out, 1568 * EMB_);
}
